// AdvancedSwinFusionModel_12111807774884
// MI455X (gfx1250) — compile-verified, matrix operand order checked by reading, not run on hardware
//
#include <hip/hip_runtime.h>
#include <hip/hip_bf16.h>
#include <math.h>

// ============================================================================
// CDNA5 (gfx1250) Swin fusion pipeline — WMMA everywhere it matters.
//   - k_gemm: 128x64 block tile, 8 waves * (32x32) each, K-step 32,
//     v_wmma_f32_16x16x32_f16, f32->f16 conversion during LDS staging,
//     vectorized (float4) global loads + register prefetch pipeline,
//     block-uniform interior fast path (no bounds checks in the hot loop),
//     optional fused LayerNorm on A, fused bias/residual/GELU/ReLU epilogue.
//   - k_attn: fully fused windowed attention per (window, head); head_dim==32
//     at every stage, N=49 padded to 64. QK^T and PV via WMMA.
// wave32 fragment layouts per CDNA5 ISA:
//   A 16x32 f16 : lane m=l&15, g=l>>4; halves 0..7 -> K=g*8+h, 8..15 -> 16+g*8+h
//   B 32x16 f16 : same with N over lanes
//   C/D 16x16 f32: lane l, vgpr i -> m = i + (l>>4)*8, n = l&15
// ============================================================================

typedef __attribute__((ext_vector_type(16))) _Float16 v16h;
typedef __attribute__((ext_vector_type(8)))  float    v8f;

#define LDH 40  // LDS row stride in halves (32 data + 8 pad)

struct F8 { float v[8]; };

// Load 8 consecutive floats src[row*K + kc .. +8) ; FAST = no bounds checks
// (two unconditional global_load_b128), else zero-fill OOB.
template<bool FAST>
__device__ __forceinline__ F8 ld8(const float* __restrict__ src, int row, int nrows,
                                  int kc, int K)
{
    F8 r;
    if (FAST || (row < nrows && kc + 8 <= K)) {
        const float4* p = reinterpret_cast<const float4*>(src + (size_t)row * K + kc);
        const float4 a = p[0], b = p[1];
        r.v[0] = a.x; r.v[1] = a.y; r.v[2] = a.z; r.v[3] = a.w;
        r.v[4] = b.x; r.v[5] = b.y; r.v[6] = b.z; r.v[7] = b.w;
    } else {
#pragma unroll
        for (int u = 0; u < 8; ++u)
            r.v[u] = (row < nrows && kc + u < K) ? src[(size_t)row * K + kc + u] : 0.0f;
    }
    return r;
}

// A-load with optional fused LayerNorm: (x - mu) * rstd * g[k] + b[k]
template<bool FAST>
__device__ __forceinline__ F8 ld8_ln(const float* __restrict__ A, int row, int M,
                                     int kc, int K,
                                     const float* __restrict__ lnG,
                                     const float* __restrict__ lnB,
                                     const float* __restrict__ stats)
{
    F8 r = ld8<FAST>(A, row, M, kc, K);
    if (stats != nullptr && (FAST || row < M)) {
        const float mu = stats[2 * row];
        const float is = stats[2 * row + 1];
        const F8 g = ld8<FAST>(lnG, 0, 1, kc, K);
        const F8 b = ld8<FAST>(lnB, 0, 1, kc, K);
#pragma unroll
        for (int u = 0; u < 8; ++u)
            r.v[u] = (r.v[u] - mu) * is * g.v[u] + b.v[u];
    }
    return r;
}

__device__ __forceinline__ void st8(_Float16* __restrict__ d, const F8& r)
{
#pragma unroll
    for (int u = 0; u < 8; ++u) d[u] = (_Float16)r.v[u];
}

// ---------------------------------------------------------------------------
// GEMM pipeline body (templated on interior-block fast path).
// ---------------------------------------------------------------------------
template<bool FAST>
__device__ __forceinline__ void gemm_body(
    const float* __restrict__ A, const float* __restrict__ Wt,
    const float* __restrict__ bias, const float* __restrict__ res,
    const float* __restrict__ lnG, const float* __restrict__ lnB,
    const float* __restrict__ stats,
    float* __restrict__ out, int M, int N, int K, int act,
    int m0, int n0, _Float16* sA, _Float16* sB)
{
    const int t    = threadIdx.x;
    const int lane = t & 31;
    const int wave = t >> 5;     // 0..7
    const int wr   = wave >> 1;  // 0..3 : 32-row band
    const int wc   = wave & 1;   // 0..1 : 32-col half

    const int lr = t >> 2;           // staging row 0..63
    const int lc = (t & 3) * 8;      // staging col 0,8,16,24

    v8f acc[2][2] = { { {}, {} }, { {}, {} } };

    const int KT = (K + 31) / 32;

    // register-prefetch pipeline: regs hold tile kt, LDS holds tile in compute
    F8 ra0 = ld8_ln<FAST>(A, m0 + lr,      M, lc, K, lnG, lnB, stats);
    F8 ra1 = ld8_ln<FAST>(A, m0 + lr + 64, M, lc, K, lnG, lnB, stats);
    F8 rb0 = ld8<FAST>(Wt, n0 + lr, N, lc, K);

    for (int kt = 0; kt < KT; ++kt) {
        st8(&sA[lr * LDH + lc], ra0);
        st8(&sA[(lr + 64) * LDH + lc], ra1);
        st8(&sB[lr * LDH + lc], rb0);
        __syncthreads();

        if (kt + 1 < KT) {
            const int kn = (kt + 1) * 32 + lc;
            ra0 = ld8_ln<FAST>(A, m0 + lr,      M, kn, K, lnG, lnB, stats);
            ra1 = ld8_ln<FAST>(A, m0 + lr + 64, M, kn, K, lnG, lnB, stats);
            rb0 = ld8<FAST>(Wt, n0 + lr, N, kn, K);
        }

        const int g8 = (lane >> 4) * 8;
        v16h af[2], bf[2];
#pragma unroll
        for (int ti = 0; ti < 2; ++ti) {
            const int ar = 32 * wr + 16 * ti + (lane & 15);
#pragma unroll
            for (int h = 0; h < 8; ++h) af[ti][h]     = sA[ar * LDH + g8 + h];
#pragma unroll
            for (int h = 0; h < 8; ++h) af[ti][8 + h] = sA[ar * LDH + 16 + g8 + h];
        }
#pragma unroll
        for (int tn = 0; tn < 2; ++tn) {
            const int br = 32 * wc + 16 * tn + (lane & 15);
#pragma unroll
            for (int h = 0; h < 8; ++h) bf[tn][h]     = sB[br * LDH + g8 + h];
#pragma unroll
            for (int h = 0; h < 8; ++h) bf[tn][8 + h] = sB[br * LDH + 16 + g8 + h];
        }
#pragma unroll
        for (int ti = 0; ti < 2; ++ti)
#pragma unroll
            for (int tn = 0; tn < 2; ++tn)
                acc[ti][tn] = __builtin_amdgcn_wmma_f32_16x16x32_f16(
                    false, af[ti], false, bf[tn], (short)0, acc[ti][tn], false, false);
        __syncthreads();
    }

    // epilogue
#pragma unroll
    for (int ti = 0; ti < 2; ++ti) {
#pragma unroll
        for (int tn = 0; tn < 2; ++tn) {
            const int col = n0 + 32 * wc + 16 * tn + (lane & 15);
            if (!FAST && col >= N) continue;
            const float bv = bias ? bias[col] : 0.0f;
#pragma unroll
            for (int i = 0; i < 8; ++i) {
                const int row = m0 + 32 * wr + 16 * ti + ((lane >> 4) << 3) + i;
                if (!FAST && row >= M) continue;
                float v = acc[ti][tn][i] + bv;
                if (res) v += res[(size_t)row * N + col];
                if (act == 1)      v = 0.5f * v * (1.0f + erff(v * 0.70710678118654752f));
                else if (act == 2) v = fmaxf(v, 0.0f);
                out[(size_t)row * N + col] = v;
            }
        }
    }
}

// GEMM: out[M,N] = act( LN?(A)[M,K] @ W[N,K]^T + bias [+ res] )
// Block tile 128x64; 8 waves, each 32x32 (4 wmma per K-step of 32).
__global__ __launch_bounds__(256)
void k_gemm(const float* __restrict__ A, const float* __restrict__ Wt,
            const float* __restrict__ bias, const float* __restrict__ res,
            const float* __restrict__ lnG, const float* __restrict__ lnB,
            const float* __restrict__ stats,
            float* __restrict__ out, int M, int N, int K, int act)
{
    __shared__ _Float16 sA[128 * LDH];
    __shared__ _Float16 sB[64 * LDH];
    const int m0 = blockIdx.y * 128;
    const int n0 = blockIdx.x * 64;
    // block-uniform dispatch: interior blocks run with zero bounds checks
    if (m0 + 128 <= M && n0 + 64 <= N && (K & 31) == 0)
        gemm_body<true >(A, Wt, bias, res, lnG, lnB, stats, out, M, N, K, act, m0, n0, sA, sB);
    else
        gemm_body<false>(A, Wt, bias, res, lnG, lnB, stats, out, M, N, K, act, m0, n0, sA, sB);
}

// ---------------------------------------------------------------------------
// Per-row LayerNorm stats: st[2r]=mean, st[2r+1]=rsqrt(var+eps)
// ---------------------------------------------------------------------------
__global__ __launch_bounds__(128)
void k_rowstats(const float* __restrict__ x, float* __restrict__ st, int C)
{
    const int row = blockIdx.x;
    const float* xr = x + (size_t)row * C;
    __shared__ float s1[128], s2[128];
    float ls = 0.f, lq = 0.f;
    for (int c = threadIdx.x; c < C; c += 128) { float v = xr[c]; ls += v; lq += v * v; }
    s1[threadIdx.x] = ls; s2[threadIdx.x] = lq;
    __syncthreads();
    for (int off = 64; off > 0; off >>= 1) {
        if ((int)threadIdx.x < off) { s1[threadIdx.x] += s1[threadIdx.x + off];
                                      s2[threadIdx.x] += s2[threadIdx.x + off]; }
        __syncthreads();
    }
    if (threadIdx.x == 0) {
        const float mean = s1[0] / C;
        const float var  = s2[0] / C - mean * mean;
        st[2 * row]     = mean;
        st[2 * row + 1] = rsqrtf(var + 1e-5f);
    }
}

// ---------------------------------------------------------------------------
// Full LayerNorm (used only after patch embed, where the LN'd value IS the
// residual stream).
// ---------------------------------------------------------------------------
__global__ __launch_bounds__(128)
void k_ln(const float* __restrict__ x, const float* __restrict__ g,
          const float* __restrict__ b, float* __restrict__ y, int C)
{
    const int row = blockIdx.x;
    const float* xr = x + (size_t)row * C;
    float* yr = y + (size_t)row * C;
    __shared__ float s1[128], s2[128];
    float ls = 0.f, lq = 0.f;
    for (int c = threadIdx.x; c < C; c += 128) { float v = xr[c]; ls += v; lq += v * v; }
    s1[threadIdx.x] = ls; s2[threadIdx.x] = lq;
    __syncthreads();
    for (int off = 64; off > 0; off >>= 1) {
        if ((int)threadIdx.x < off) { s1[threadIdx.x] += s1[threadIdx.x + off];
                                      s2[threadIdx.x] += s2[threadIdx.x + off]; }
        __syncthreads();
    }
    const float mean = s1[0] / C;
    const float var  = s2[0] / C - mean * mean;
    const float inv  = rsqrtf(var + 1e-5f);
    for (int c = threadIdx.x; c < C; c += 128)
        yr[c] = (xr[c] - mean) * inv * g[c] + b[c];
}

// ---------------------------------------------------------------------------
// Fused windowed attention: one block per (head, window, image).
// ---------------------------------------------------------------------------
__device__ __forceinline__ int reg7(int h, int H) {
    return (h < H - 7) ? 0 : ((h < H - 3) ? 1 : 2);
}

__global__ __launch_bounds__(128)
void k_attn(const float* __restrict__ qkv, const float* __restrict__ rpb,
            float* __restrict__ attout, int H, int W, int C, int heads, int shift)
{
    const int head = blockIdx.x;
    const int wIdx = blockIdx.y;
    const int bt   = blockIdx.z;
    const int nWw  = W / 7;
    const int wh   = wIdx / nWw;
    const int wwi  = wIdx % nWw;
    const int L    = H * W;
    const int t    = threadIdx.x;
    const int lane = t & 31;
    const int wv   = t >> 5;

    __shared__ _Float16 sq[64 * 32];
    __shared__ _Float16 sk[64 * 32];
    __shared__ _Float16 sv[64 * 32];
    __shared__ float    ss[64 * 64];
    __shared__ _Float16 sp[64 * 64];

    // ---- gather Q(scaled)/K/V as f16, vectorized; zero-pad rows 49..63 ----
    {
        const int r  = t >> 1;
        const int c0 = (t & 1) * 16;
        const float qs = 0.17677669529663688f; // 32^-0.5
        if (r < 49) {
            const int i  = r / 7, j = r % 7;
            const int sh = (wh * 7 + i + shift) % H;
            const int sw = (wwi * 7 + j + shift) % W;
            const size_t base =
                ((size_t)bt * L + (size_t)sh * W + sw) * (size_t)(3 * C) + head * 32 + c0;
            const float4* pq = reinterpret_cast<const float4*>(qkv + base);
            const float4* pk = reinterpret_cast<const float4*>(qkv + base + C);
            const float4* pv = reinterpret_cast<const float4*>(qkv + base + 2 * C);
#pragma unroll
            for (int u = 0; u < 4; ++u) {
                const float4 q4 = pq[u], k4 = pk[u], v4 = pv[u];
                const int d = r * 32 + c0 + u * 4;
                sq[d + 0] = (_Float16)(q4.x * qs); sq[d + 1] = (_Float16)(q4.y * qs);
                sq[d + 2] = (_Float16)(q4.z * qs); sq[d + 3] = (_Float16)(q4.w * qs);
                sk[d + 0] = (_Float16)k4.x; sk[d + 1] = (_Float16)k4.y;
                sk[d + 2] = (_Float16)k4.z; sk[d + 3] = (_Float16)k4.w;
                sv[d + 0] = (_Float16)v4.x; sv[d + 1] = (_Float16)v4.y;
                sv[d + 2] = (_Float16)v4.z; sv[d + 3] = (_Float16)v4.w;
            }
        } else {
#pragma unroll
            for (int u = 0; u < 16; ++u) {
                const int d = r * 32 + c0 + u;
                sq[d] = (_Float16)0.f; sk[d] = (_Float16)0.f; sv[d] = (_Float16)0.f;
            }
        }
    }
    __syncthreads();

    // ---- S = Q @ K^T (64x64, K=32) ----
    {
        const int ar = 16 * wv + (lane & 15);
        const int g8 = (lane >> 4) * 8;
        v16h af;
#pragma unroll
        for (int h = 0; h < 8; ++h) af[h]     = sq[ar * 32 + g8 + h];
#pragma unroll
        for (int h = 0; h < 8; ++h) af[8 + h] = sq[ar * 32 + 16 + g8 + h];

        v8f acc[4];
#pragma unroll
        for (int tn = 0; tn < 4; ++tn) {
            const int br = tn * 16 + (lane & 15);
            v16h bf;
#pragma unroll
            for (int h = 0; h < 8; ++h) bf[h]     = sk[br * 32 + g8 + h];
#pragma unroll
            for (int h = 0; h < 8; ++h) bf[8 + h] = sk[br * 32 + 16 + g8 + h];
            v8f cz = {};
            acc[tn] = __builtin_amdgcn_wmma_f32_16x16x32_f16(
                false, af, false, bf, (short)0, cz, false, false);
        }

        // scores + rel-pos bias + shift mask (computed arithmetically)
#pragma unroll
        for (int tn = 0; tn < 4; ++tn) {
            const int m  = tn * 16 + (lane & 15);
            const int i2 = m / 7, j2 = m % 7;
#pragma unroll
            for (int i = 0; i < 8; ++i) {
                const int n = 16 * wv + ((lane >> 4) << 3) + i;
                float val = -1e30f;
                if (n < 49 && m < 49) {
                    const int i1 = n / 7, j1 = n % 7;
                    const int ridx = (i1 - i2 + 6) * 13 + (j1 - j2 + 6);
                    val = acc[tn][i] + rpb[ridx * heads + head];
                    if (shift > 0) {
                        const int rn = reg7(wh * 7 + i1, H) * 3 + reg7(wwi * 7 + j1, W);
                        const int rm = reg7(wh * 7 + i2, H) * 3 + reg7(wwi * 7 + j2, W);
                        if (rn != rm) val -= 100.0f;
                    }
                }
                ss[n * 64 + m] = val;
            }
        }
    }
    __syncthreads();

    // ---- softmax rows ----
    if (t < 64) {
        float mx = -1e30f;
        for (int m = 0; m < 64; ++m) mx = fmaxf(mx, ss[t * 64 + m]);
        float sum = 0.0f;
        for (int m = 0; m < 64; ++m) {
            float e = __expf(ss[t * 64 + m] - mx);
            sum += e;
            ss[t * 64 + m] = e;
        }
        const float inv = 1.0f / sum;
        for (int m = 0; m < 64; ++m) sp[t * 64 + m] = (_Float16)(ss[t * 64 + m] * inv);
    }
    __syncthreads();

    // ---- O = P @ V (64x32, K=64 in two steps) + scatter (undo roll) ----
    {
        const int ar = 16 * wv + (lane & 15);
        const int g8 = (lane >> 4) * 8;
        v8f acc[2] = { {}, {} };
#pragma unroll
        for (int kk = 0; kk < 64; kk += 32) {
            v16h af;
#pragma unroll
            for (int h = 0; h < 8; ++h) af[h]     = sp[ar * 64 + kk + g8 + h];
#pragma unroll
            for (int h = 0; h < 8; ++h) af[8 + h] = sp[ar * 64 + kk + 16 + g8 + h];
#pragma unroll
            for (int tn = 0; tn < 2; ++tn) {
                const int nn = tn * 16 + (lane & 15);
                v16h bf;
#pragma unroll
                for (int h = 0; h < 8; ++h) bf[h]     = sv[(kk + g8 + h) * 32 + nn];
#pragma unroll
                for (int h = 0; h < 8; ++h) bf[8 + h] = sv[(kk + 16 + g8 + h) * 32 + nn];
                acc[tn] = __builtin_amdgcn_wmma_f32_16x16x32_f16(
                    false, af, false, bf, (short)0, acc[tn], false, false);
            }
        }
#pragma unroll
        for (int tn = 0; tn < 2; ++tn) {
            const int d = tn * 16 + (lane & 15);
#pragma unroll
            for (int i = 0; i < 8; ++i) {
                const int n = 16 * wv + ((lane >> 4) << 3) + i;
                if (n < 49) {
                    const int i1 = n / 7, j1 = n % 7;
                    const int sh = (wh * 7 + i1 + shift) % H;
                    const int sw = (wwi * 7 + j1 + shift) % W;
                    attout[((size_t)bt * L + (size_t)sh * W + sw) * C + head * 32 + d] =
                        acc[tn][i];
                }
            }
        }
    }
}

// ---------------------------------------------------------------------------
// im2col for non-overlapping 4x4 patches: (16,3,224,224) -> (16*3136, 48)
// ---------------------------------------------------------------------------
__global__ void k_im2col(const float* __restrict__ xir, const float* __restrict__ xvis,
                         float* __restrict__ out, size_t total)
{
    size_t idx = (size_t)blockIdx.x * 256 + threadIdx.x;
    if (idx >= total) return;
    const int col = (int)(idx % 48);
    const size_t tok = idx / 48;
    const int p  = (int)(tok % 3136);
    const int bt = (int)(tok / 3136);
    const int ci = col / 16, rem = col % 16, dy = rem / 4, dx = rem % 4;
    const int ph = p / 56, pw = p % 56;
    const float* src = (bt < 8) ? xir : xvis;
    const int b = bt & 7;
    out[idx] = src[(((size_t)b * 3 + ci) * 224 + (ph * 4 + dy)) * 224 + (pw * 4 + dx)];
}

// ir = ir + vis ; vis = vis + ir_new
__global__ void k_crossadd(float* __restrict__ x, size_t half)
{
    size_t i = (size_t)blockIdx.x * 256 + threadIdx.x;
    if (i >= half) return;
    float a = x[i], b = x[half + i];
    x[i] = a + b;
    x[half + i] = a + 2.0f * b;
}

// patch-merge gather: (Bt,H,W,C) -> (Bt, H/2*W/2, 4C) with [00,10,01,11] order
__global__ void k_merge_gather(const float* __restrict__ x, float* __restrict__ y,
                               int H, int W, int C, size_t total)
{
    size_t idx = (size_t)blockIdx.x * 256 + threadIdx.x;
    if (idx >= total) return;
    const int C4 = 4 * C;
    const int c4 = (int)(idx % C4);
    const size_t tok = idx / C4;
    const int W2 = W / 2, H2 = H / 2;
    const int w2 = (int)(tok % W2);
    const int h2 = (int)((tok / W2) % H2);
    const int bt = (int)(tok / ((size_t)W2 * H2));
    const int q = c4 / C, c = c4 % C;
    const int hh = 2 * h2 + ((q == 1 || q == 3) ? 1 : 0);
    const int ww = 2 * w2 + ((q >= 2) ? 1 : 0);
    y[idx] = x[((size_t)bt * H * W + (size_t)hh * W + ww) * C + c];
}

// (B,L,C) tokens -> NCHW
__global__ void k_nchw(const float* __restrict__ x, float* __restrict__ y,
                       int Cc, int H, int W, size_t total)
{
    size_t idx = (size_t)blockIdx.x * 256 + threadIdx.x;
    if (idx >= total) return;
    const int w = (int)(idx % W);
    const int h = (int)((idx / W) % H);
    const int c = (int)((idx / ((size_t)W * H)) % Cc);
    const int b = (int)(idx / ((size_t)W * H * Cc));
    y[idx] = x[((size_t)b * (H * W) + h * W + w) * Cc + c];
}

// ConvTranspose2d k2 s2 fused with BN(eval) + ReLU. Kw layout (Ci,Co,2,2).
__global__ void k_convt2(const float* __restrict__ x, const float* __restrict__ Kw,
                         const float* __restrict__ bias,
                         const float* __restrict__ g, const float* __restrict__ bb,
                         const float* __restrict__ rm, const float* __restrict__ rv,
                         float* __restrict__ y, int Ci, int Co, int H, int W, size_t total)
{
    size_t idx = (size_t)blockIdx.x * 256 + threadIdx.x;
    if (idx >= total) return;
    const int OW = 2 * W, OH = 2 * H;
    const int ow = (int)(idx % OW);
    const int oh = (int)((idx / OW) % OH);
    const int o  = (int)((idx / ((size_t)OW * OH)) % Co);
    const int b  = (int)(idx / ((size_t)OW * OH * Co));
    const int h = oh >> 1, d = oh & 1, w = ow >> 1, k2 = ow & 1;
    float s = bias[o];
    const float* xp = x + ((size_t)b * Ci) * H * W + (size_t)h * W + w;
    const float* kp = Kw + (size_t)o * 4 + d * 2 + k2;
    for (int c = 0; c < Ci; ++c)
        s += xp[(size_t)c * H * W] * kp[(size_t)c * Co * 4];
    const float sc = g[o] * rsqrtf(rv[o] + 1e-5f);
    float v = s * sc + (bb[o] - rm[o] * sc);
    y[idx] = fmaxf(v, 0.0f);
}

// Conv2d (ks x ks, pad), Kw layout (Co,Ci,ks,ks). mode: 0 plain, 1 BN+ReLU.
__global__ void k_conv2d(const float* __restrict__ x, const float* __restrict__ Kw,
                         const float* __restrict__ bias,
                         const float* __restrict__ g, const float* __restrict__ bb,
                         const float* __restrict__ rm, const float* __restrict__ rv,
                         float* __restrict__ y, int Ci, int Co, int H, int W,
                         int ks, int pad, int mode, size_t total)
{
    size_t idx = (size_t)blockIdx.x * 256 + threadIdx.x;
    if (idx >= total) return;
    const int ow = (int)(idx % W);
    const int oh = (int)((idx / W) % H);
    const int o  = (int)((idx / ((size_t)W * H)) % Co);
    const int b  = (int)(idx / ((size_t)W * H * Co));
    float s = bias[o];
    for (int c = 0; c < Ci; ++c)
        for (int ky = 0; ky < ks; ++ky) {
            const int ih = oh - pad + ky;
            if (ih < 0 || ih >= H) continue;
            for (int kx = 0; kx < ks; ++kx) {
                const int iw = ow - pad + kx;
                if (iw < 0 || iw >= W) continue;
                s += x[(((size_t)b * Ci + c) * H + ih) * W + iw] *
                     Kw[(((size_t)o * Ci + c) * ks + ky) * ks + kx];
            }
        }
    if (mode == 1) {
        const float sc = g[o] * rsqrtf(rv[o] + 1e-5f);
        s = fmaxf(s * sc + (bb[o] - rm[o] * sc), 0.0f);
    }
    y[idx] = s;
}

__global__ void k_final(const float* __restrict__ fused, const float* __restrict__ detail,
                        float* __restrict__ out, size_t total)
{
    size_t idx = (size_t)blockIdx.x * 256 + threadIdx.x;
    if (idx >= total) return;
    const float v = fused[idx] + 0.1f * detail[idx];
    out[idx] = 1.0f / (1.0f + __expf(-v));
}

// ============================================================================
// Host orchestration
// ============================================================================
extern "C" void kernel_launch(void* const* d_in, const int* in_sizes, int n_in,
                              void* d_out, int out_size, void* d_ws, size_t ws_size,
                              hipStream_t stream)
{
    (void)in_sizes; (void)n_in; (void)out_size; (void)ws_size;
    auto f = [&](int i) { return (const float*)d_in[i]; };
    int ip = 0;

    const float* x_ir  = f(ip++);
    const float* x_vis = f(ip++);
    const float* irW = f(ip++); const float* irB = f(ip++);
    const float* irG = f(ip++); const float* irN = f(ip++);
    const float* vsW = f(ip++); const float* vsB = f(ip++);
    const float* vsG = f(ip++); const float* vsN = f(ip++);
    // 12 blocks x 13: n1g n1b qkvW qkvb rpb projW projb n2g n2b fc1W fc1b fc2W fc2b
    const float* blk[12][13];
    for (int i = 0; i < 12; ++i)
        for (int j = 0; j < 13; ++j) blk[i][j] = f(ip++);
    const float *mgN[3], *mgB[3], *mgW[3];
    for (int i = 0; i < 3; ++i) { mgN[i] = f(ip++); mgB[i] = f(ip++); mgW[i] = f(ip++); }
    const float *uw[4], *ub[4], *ug[3], *ubt[3], *urm[3], *urv[3];
    for (int i = 0; i < 3; ++i) {
        uw[i] = f(ip++); ub[i] = f(ip++);
        ug[i] = f(ip++); ubt[i] = f(ip++); urm[i] = f(ip++); urv[i] = f(ip++);
    }
    uw[3] = f(ip++); ub[3] = f(ip++);
    const float* dw1 = f(ip++); const float* db1 = f(ip++);
    const float* dg  = f(ip++); const float* dbt = f(ip++);
    const float* drm = f(ip++); const float* drv = f(ip++);
    const float* dw2 = f(ip++); const float* db2 = f(ip++);

    // workspace (floats): X | XN | QKV | ATT | HB  (+XN doubles as LN stats buf)
    float* wsf = (float*)d_ws;
    float* X   = wsf;
    float* XN  = wsf + 5000000;
    float* QKV = wsf + 10000000;
    float* ATT = wsf + 24500000;
    float* HB  = wsf + 29500000;
    float* ST  = XN;  // per-row LN stats (2 floats/row), only live between stats+gemm

    auto gemm = [&](const float* A, const float* Wt, const float* bias, const float* res,
                    const float* lnG, const float* lnB, const float* stats,
                    float* out, int M, int N, int K, int act) {
        dim3 g((unsigned)((N + 63) / 64), (unsigned)((M + 127) / 128));
        k_gemm<<<g, 256, 0, stream>>>(A, Wt, bias, res, lnG, lnB, stats,
                                      out, M, N, K, act);
    };
    auto eblk = [](size_t n) { return (unsigned)((n + 255) / 256); };

    // ---- patch embed: im2col -> WMMA GEMM -> LN (LN output IS the stream) ----
    {
        const size_t itot = (size_t)16 * 3136 * 48;
        k_im2col<<<eblk(itot), 256, 0, stream>>>(x_ir, x_vis, HB, itot);
        const size_t hs = (size_t)8 * 3136;
        gemm(HB,           irW, irB, nullptr, nullptr, nullptr, nullptr,
             QKV,           (int)hs, 96, 48, 0);
        gemm(HB + hs * 48, vsW, vsB, nullptr, nullptr, nullptr, nullptr,
             QKV + hs * 96, (int)hs, 96, 48, 0);
        k_ln<<<(unsigned)hs, 128, 0, stream>>>(QKV,           irG, irN, X,           96);
        k_ln<<<(unsigned)hs, 128, 0, stream>>>(QKV + hs * 96, vsG, vsN, X + hs * 96, 96);
    }

    // ---- swin stages (ir+vis batched as Bt=16, shared params) ----
    int H = 56, W = 56, C = 96, bi = 0;
    const int depths[4]   = {2, 2, 6, 2};
    const int headsArr[4] = {3, 6, 12, 24};
    for (int s = 0; s < 4; ++s) {
        const int L = H * W, T = 16 * L, heads = headsArr[s];
        const int shift_base = (H <= 7) ? 0 : 3;
        for (int j = 0; j < depths[s]; ++j, ++bi) {
            const float* const* p = blk[bi];
            const int shift = (j & 1) ? shift_base : 0;
            // qkv = LN1(X) @ qkvW^T + b   (LN fused into GEMM A-staging)
            k_rowstats<<<(unsigned)T, 128, 0, stream>>>(X, ST, C);
            gemm(X, p[2], p[3], nullptr, p[0], p[1], ST, QKV, T, 3 * C, C, 0);
            dim3 ag((unsigned)heads, (unsigned)((H / 7) * (W / 7)), 16);
            k_attn<<<ag, 128, 0, stream>>>(QKV, p[4], ATT, H, W, C, heads, shift);
            // X = X + ATT @ projW^T + b
            gemm(ATT, p[5], p[6], X, nullptr, nullptr, nullptr, X, T, C, C, 0);
            // HB = gelu( LN2(X) @ fc1W^T + b )
            k_rowstats<<<(unsigned)T, 128, 0, stream>>>(X, ST, C);
            gemm(X, p[9], p[10], nullptr, p[7], p[8], ST, HB, T, 4 * C, C, 1);
            // X = X + HB @ fc2W^T + b
            gemm(HB, p[11], p[12], X, nullptr, nullptr, nullptr, X, T, C, 4 * C, 0);
        }
        const size_t half = (size_t)8 * L * C;
        k_crossadd<<<eblk(half), 256, 0, stream>>>(X, half);
        if (s < 3) {
            const int Tn = 16 * (H / 2) * (W / 2);
            const size_t gtot = (size_t)Tn * 4 * C;
            k_merge_gather<<<eblk(gtot), 256, 0, stream>>>(X, HB, H, W, C, gtot);
            k_rowstats<<<(unsigned)Tn, 128, 0, stream>>>(HB, ST, 4 * C);
            gemm(HB, mgW[s], nullptr, nullptr, mgN[s], mgB[s], ST, X, Tn, 2 * C, 4 * C, 0);
            H /= 2; W /= 2; C *= 2;
        }
    }

    // ---- decoder (ir half of X): (8,49,768) -> (8,3,56,56) ----
    {
        const size_t ntot = (size_t)8 * 768 * 49;
        k_nchw<<<eblk(ntot), 256, 0, stream>>>(X, XN, 768, 7, 7, ntot);
        const size_t c1 = (size_t)8 * 384 * 14 * 14;
        k_convt2<<<eblk(c1), 256, 0, stream>>>(XN, uw[0], ub[0], ug[0], ubt[0], urm[0], urv[0],
                                               HB, 768, 384, 7, 7, c1);
        const size_t c2 = (size_t)8 * 192 * 28 * 28;
        k_convt2<<<eblk(c2), 256, 0, stream>>>(HB, uw[1], ub[1], ug[1], ubt[1], urm[1], urv[1],
                                               XN, 384, 192, 14, 14, c2);
        const size_t c3 = (size_t)8 * 96 * 56 * 56;
        k_convt2<<<eblk(c3), 256, 0, stream>>>(XN, uw[2], ub[2], ug[2], ubt[2], urm[2], urv[2],
                                               HB, 192, 96, 28, 28, c3);
        const size_t c4 = (size_t)8 * 3 * 56 * 56;
        k_conv2d<<<eblk(c4), 256, 0, stream>>>(HB, uw[3], ub[3],
                                               nullptr, nullptr, nullptr, nullptr,
                                               QKV, 96, 3, 56, 56, 1, 0, 0, c4);   // fused
        const size_t c5 = (size_t)8 * 16 * 56 * 56;
        k_conv2d<<<eblk(c5), 256, 0, stream>>>(QKV, dw1, db1, dg, dbt, drm, drv,
                                               ATT, 3, 16, 56, 56, 3, 1, 1, c5);
        k_conv2d<<<eblk(c4), 256, 0, stream>>>(ATT, dw2, db2,
                                               nullptr, nullptr, nullptr, nullptr,
                                               XN, 16, 3, 56, 56, 3, 1, 0, c4);    // detail
        k_final<<<eblk(c4), 256, 0, stream>>>(QKV, XN, (float*)d_out, c4);
    }
}
